// SecondVoxelTower_48086453846505
// MI455X (gfx1250) — matrix operand order repaired and checked by reading, not compiled, not run on hardware
//
#include <hip/hip_runtime.h>
#include <hip/hip_bf16.h>
#include <hip/hip_fp16.h>

typedef __attribute__((ext_vector_type(16))) _Float16 v16h;
typedef __attribute__((ext_vector_type(8)))  _Float16 v8h;
typedef __attribute__((ext_vector_type(8)))  float    v8f;

// ---------------------------------------------------------------- utilities

__global__ void k_zero(float* p, long n) {
    long i = (long)blockIdx.x * blockDim.x + threadIdx.x;
    long stride = (long)gridDim.x * blockDim.x;
    for (; i < n; i += stride) p[i] = 0.0f;
}

// ---------------------------------------------------------------- voxelize

__global__ void k_voxelize(const float* __restrict__ pts, float* __restrict__ acc,
                           int B, int N) {
    long total = (long)B * N;
    long gid = (long)blockIdx.x * blockDim.x + threadIdx.x;
    long stride = (long)gridDim.x * blockDim.x;
    for (; gid < total; gid += stride) {
        const float* p = pts + gid * 3;
        float x = p[0], y = p[1], z = p[2];
        // PC_MIN = (-0.5,-0.5,0), VS = 0.0125 -> (v - min) * 80
        int ix = min(max((int)floorf((x + 0.5f) * 80.0f), 0), 79);
        int iy = min(max((int)floorf((y + 0.5f) * 80.0f), 0), 79);
        int iz = min(max((int)floorf(z * 80.0f), 0), 79);
        int b = (int)(gid / N);
        long flat = (((long)b * 80 + iz) * 80 + iy) * 80 + ix;
        atomicAdd(acc + flat * 4 + 0, x);
        atomicAdd(acc + flat * 4 + 1, y);
        atomicAdd(acc + flat * 4 + 2, z);
        atomicAdd(acc + flat * 4 + 3, 1.0f);
    }
}

// per voxel: mean xyz -> channel-last f16 with C padded to 32 (pad = 0)
__global__ void k_mean(const float* __restrict__ acc, _Float16* __restrict__ dst,
                       long nvox) {
    long gid = (long)blockIdx.x * blockDim.x + threadIdx.x;
    long stride = (long)gridDim.x * blockDim.x;
    for (; gid < nvox; gid += stride) {
        const float* a = acc + gid * 4;
        float inv = 1.0f / fmaxf(a[3], 1.0f);
        _Float16* d = dst + gid * 32;
        d[0] = (_Float16)(a[0] * inv);
        d[1] = (_Float16)(a[1] * inv);
        d[2] = (_Float16)(a[2] * inv);
        #pragma unroll
        for (int j = 3; j < 32; ++j) d[j] = (_Float16)0.0f;
    }
}

// ---------------------------------------------------------------- weight pack
// Packs f32 OIDHW weights into the exact per-lane WMMA B-fragment layout:
//   out[(((tap*kch + kc)*ntiles + ct)*32 + lane)*16 + j]
// B (32x16, KxN) layout per ISA 7.12.2: lane n(0..15) = col n with K={0..7,16..23},
// lane 16+n = col n with K={8..15,24..31}, sequential pairs per VGPR.
__global__ void k_pack(const float* __restrict__ w, _Float16* __restrict__ out,
                       int Cin, int Cinpad, int Cout, int ntiles,
                       int KD, int KH, int KW) {
    int kch = Cinpad >> 5;
    long total = (long)KD * KH * KW * kch * ntiles * 512;
    long idx = (long)blockIdx.x * blockDim.x + threadIdx.x;
    long stride = (long)gridDim.x * blockDim.x;
    for (; idx < total; idx += stride) {
        int j = (int)(idx & 15);
        int lane = (int)((idx >> 4) & 31);
        long t = idx >> 9;
        int ct = (int)(t % ntiles); t /= ntiles;
        int kc = (int)(t % kch);
        int tap = (int)(t / kch);
        int kx = tap % KW;
        int ky = (tap / KW) % KH;
        int kz = tap / (KW * KH);
        int n = lane & 15, khalf = lane >> 4;
        int k = kc * 32 + (j < 8 ? khalf * 8 + j : 16 + khalf * 8 + (j - 8));    // B-fragment K ordering
        int co = ct * 16 + n;
        float v = 0.0f;
        if (co < Cout && k < Cin)
            v = w[((((long)co * Cin + k) * KD + kz) * KH + ky) * KW + kx];
        out[idx] = (_Float16)v;
    }
}

// ---------------------------------------------------------------- implicit-GEMM conv
// One wave per 16 (output voxels) x (16*NB) (output channels) tile, KCH
// K-chunks of 32 input channels per tap. An empty-asm scheduling barrier
// pins the A fragment + all NB*KCH B fragments live at once: the loads go
// out as one clause with a single wait, then NB*KCH independent WMMAs issue
// back-to-back (covering the f16-WMMA RAW hazard without v_nops). A-global
// traffic drops by NB. Activations channel-last f16, C padded to mult of 32.
template <int NB, int KCH>
__global__ void __launch_bounds__(32) k_conv(
    const _Float16* __restrict__ src, _Float16* __restrict__ dst,
    const _Float16* __restrict__ wf,
    int ID, int IH, int IW, int OD, int OH, int OW,
    int Coutpad, int ntiles,
    int sz, int sy, int sx, int pz, int py, int px,
    int KD, int KH, int KW)
{
    const int Cinpad = KCH * 32;
    const int lane  = threadIdx.x;
    const int n     = lane & 15;
    const int khalf = lane >> 4;
    const int tileM = blockIdx.x;
    const int ct0   = blockIdx.y * NB;
    const int b     = blockIdx.z;
    const int nvox  = OD * OH * OW;

    // A-matrix row for this lane (lanes l and l+16 share row l, different K halves)
    const int vmA = tileM * 16 + n;
    const bool rowOK = vmA < nvox;
    int oz = 0, oy = 0, ox = 0;
    if (rowOK) {
        ox = vmA % OW;
        int t = vmA / OW;
        oy = t % OH;
        oz = t / OH;
    }

    v8f acc[NB];
    #pragma unroll
    for (int q = 0; q < NB; ++q) acc[q] = (v8f){};

    int tap = 0;
    for (int kz = 0; kz < KD; ++kz) {
        int iz = oz * sz - pz + kz;
        for (int ky = 0; ky < KH; ++ky) {
            int iy = oy * sy - py + ky;
            for (int kx = 0; kx < KW; ++kx, ++tap) {
                int ix = ox * sx - px + kx;
                bool valid = rowOK &&
                             (unsigned)iz < (unsigned)ID &&
                             (unsigned)iy < (unsigned)IH &&
                             (unsigned)ix < (unsigned)IW;
                const _Float16* ap = src +
                    ((((long)b * ID + iz) * IH + iy) * IW + ix) * (long)Cinpad +
                    khalf * 8;
                const _Float16* wp = wf +
                    (((long)tap * KCH * ntiles + ct0) * 32 + lane) * 16;

                // A 16x32 f16 layout (ISA 7.12.2): lane<16 K={0..7,16..23},
                // lane>=16 K={8..15,24..31}
                v16h A[KCH];
                #pragma unroll
                for (int kc = 0; kc < KCH; ++kc) {
                    v8h lo = {}, hi = {};
                    if (valid) {
                        lo = *(const v8h*)(ap + kc * 32);
                        hi = *(const v8h*)(ap + kc * 32 + 16);
                    }
                    #pragma unroll
                    for (int j = 0; j < 8; ++j) {
                        A[kc][j] = lo[j];
                        A[kc][8 + j] = hi[j];
                    }
                }
                v16h Bm[KCH][NB];
                #pragma unroll
                for (int kc = 0; kc < KCH; ++kc)
                    #pragma unroll
                    for (int q = 0; q < NB; ++q)
                        Bm[kc][q] = *(const v16h*)(wp +
                            ((long)kc * ntiles * 32 + (long)q * 32) * 16);

                // scheduling barrier: force all fragments live -> one load
                // clause + single wait, then uninterrupted WMMA burst
                if constexpr (KCH == 1 && NB == 2)
                    asm volatile("" : "+v"(A[0]), "+v"(Bm[0][0]), "+v"(Bm[0][1]));
                else if constexpr (KCH == 1 && NB == 4)
                    asm volatile("" : "+v"(A[0]), "+v"(Bm[0][0]), "+v"(Bm[0][1]),
                                      "+v"(Bm[0][2]), "+v"(Bm[0][3]));
                else if constexpr (KCH == 2 && NB == 4)
                    asm volatile("" : "+v"(A[0]), "+v"(A[1]),
                                      "+v"(Bm[0][0]), "+v"(Bm[0][1]),
                                      "+v"(Bm[0][2]), "+v"(Bm[0][3]),
                                      "+v"(Bm[1][0]), "+v"(Bm[1][1]),
                                      "+v"(Bm[1][2]), "+v"(Bm[1][3]));

                #pragma unroll
                for (int kc = 0; kc < KCH; ++kc)
                    #pragma unroll
                    for (int q = 0; q < NB; ++q)
                        acc[q] = __builtin_amdgcn_wmma_f32_16x16x32_f16(
                            false, A[kc], false, Bm[kc][q], (short)0, acc[q],
                            false, false);
            }
        }
    }

    // C/D layout: VGPR r holds M = r + 8*(lane>=16), N = lane&15. ReLU + f16 store.
    #pragma unroll
    for (int q = 0; q < NB; ++q) {
        const long obase = ((long)b * nvox) * Coutpad + (long)(ct0 + q) * 16 + n;
        #pragma unroll
        for (int r = 0; r < 8; ++r) {
            int vox = tileM * 16 + r + 8 * khalf;
            if (vox < nvox) {
                float v = fmaxf(acc[q][r], 0.0f);
                dst[obase + (long)vox * Coutpad] = (_Float16)v;
            }
        }
    }
}

// ---------------------------------------------------------------- outputs
// act: [B, 400, 4096] f16 (channel-last == feats layout). Emits bf16 feats,
// bf16 valid flags, and accumulates per-batch valid counts.
__global__ void __launch_bounds__(256) k_final(
    const _Float16* __restrict__ act, __hip_bfloat16* __restrict__ out,
    int* counters, int C, int nvox, long feats, long out_size)
{
    __shared__ float red[256];
    const int bv = blockIdx.x;          // b*nvox + vox
    const int t  = threadIdx.x;
    const _Float16* a = act + (long)bv * C;
    float s = 0.0f;
    for (int c = t; c < C; c += 256) {
        float v = (float)a[c];
        s += v;
        long oi = (long)bv * C + c;
        if (oi < out_size) out[oi] = __float2bfloat16(v);
    }
    red[t] = s;
    __syncthreads();
    for (int w = 128; w > 0; w >>= 1) {
        if (t < w) red[t] += red[t + w];
        __syncthreads();
    }
    if (t == 0) {
        bool valid = red[0] > 0.0f;
        long oi = feats + bv;
        if (oi < out_size) out[oi] = __float2bfloat16(valid ? 1.0f : 0.0f);
        if (valid) atomicAdd(&counters[bv / nvox], 1);
    }
}

__global__ void k_offsets(const int* __restrict__ counters,
                          __hip_bfloat16* __restrict__ out,
                          long base, int B, long out_size) {
    int b = threadIdx.x;
    if (b < B) {
        long oi = base + b;
        if (oi < out_size) out[oi] = __float2bfloat16((float)(counters[b] + 1));
    }
}

// ---------------------------------------------------------------- launch

extern "C" void kernel_launch(void* const* d_in, const int* in_sizes, int n_in,
                              void* d_out, int out_size, void* d_ws, size_t ws_size,
                              hipStream_t stream) {
    (void)in_sizes; (void)n_in; (void)ws_size;
    const float* pts = (const float*)d_in[0];
    const int  B = 2, N = 100000, G = 80;
    const long G3 = (long)G * G * G;

    char* ws = (char*)d_ws;
    float* vox = (float*)ws;                                   // [B*G3*4] f32
    const long VOX_BYTES = (long)B * G3 * 4 * sizeof(float);   // 16,384,000
    int* counters = (int*)(ws + VOX_BYTES);                    // inside 256B pad
    const long OFF_ACTA  = VOX_BYTES + 256;
    const long ACT_BYTES = (long)B * G3 * 32 * sizeof(_Float16); // 65,536,000
    _Float16* actA = (_Float16*)(ws + OFF_ACTA);
    _Float16* actB = (_Float16*)(ws + OFF_ACTA + ACT_BYTES);
    char* wbase = ws + OFF_ACTA + 2 * ACT_BYTES;

    // zero voxel accumulators + counters
    k_zero<<<2048, 256, 0, stream>>>((float*)ws, B * G3 * 4 + 64);
    k_voxelize<<<1024, 256, 0, stream>>>(pts, vox, B, N);
    k_mean<<<2048, 256, 0, stream>>>(vox, actA, (long)B * G3);

    struct L { int widx, ID, IH, IW, OD, OH, OW, Cin, Cinpad, Cout, Coutpad,
                   sz, sy, sx, pz, py, px, KD, KH, KW; };
    const L Ls[12] = {
        { 1, 80,80,80, 80,80,80,  3,32,   16,32,   1,1,1, 1,1,1, 3,3,3},
        { 2, 80,80,80, 80,80,80, 16,32,   16,32,   1,1,1, 1,1,1, 3,3,3},
        { 3, 80,80,80, 40,40,40, 16,32,   32,32,   2,2,2, 1,1,1, 3,3,3},
        { 4, 40,40,40, 40,40,40, 32,32,   32,32,   1,1,1, 1,1,1, 3,3,3},
        { 5, 40,40,40, 40,40,40, 32,32,   32,32,   1,1,1, 1,1,1, 3,3,3},
        { 6, 40,40,40, 20,20,20, 32,32,   64,64,   2,2,2, 1,1,1, 3,3,3},
        { 7, 20,20,20, 20,20,20, 64,64,   64,64,   1,1,1, 1,1,1, 3,3,3},
        { 8, 20,20,20, 20,20,20, 64,64,   64,64,   1,1,1, 1,1,1, 3,3,3},
        { 9, 20,20,20,  9,10,10, 64,64,   64,64,   2,2,2, 0,1,1, 3,3,3},
        {10,  9,10,10,  9,10,10, 64,64,   64,64,   1,1,1, 1,1,1, 3,3,3},
        {11,  9,10,10,  9,10,10, 64,64,   64,64,   1,1,1, 1,1,1, 3,3,3},
        {12,  9,10,10,  4,10,10, 64,64, 4096,4096, 2,1,1, 0,0,0, 3,1,1},
    };

    _Float16* src = actA;
    _Float16* dst = actB;
    long woff = 0;
    for (int i = 0; i < 12; ++i) {
        const L& l = Ls[i];
        const int ntiles = l.Coutpad / 16;
        const int kch = l.Cinpad / 32;
        _Float16* wf = (_Float16*)(wbase + woff);
        const long welems = (long)l.KD * l.KH * l.KW * l.Cinpad * l.Coutpad;
        woff = (woff + welems * (long)sizeof(_Float16) + 255) & ~255L;

        k_pack<<<(int)((welems + 255) / 256), 256, 0, stream>>>(
            (const float*)d_in[l.widx], wf, l.Cin, l.Cinpad, l.Cout, ntiles,
            l.KD, l.KH, l.KW);

        const int nvox = l.OD * l.OH * l.OW;
        const int mtiles = (nvox + 15) / 16;
        if (kch == 1 && (ntiles % 4) != 0) {
            dim3 grid(mtiles, ntiles / 2, B);
            k_conv<2, 1><<<grid, 32, 0, stream>>>(
                src, dst, wf,
                l.ID, l.IH, l.IW, l.OD, l.OH, l.OW, l.Coutpad, ntiles,
                l.sz, l.sy, l.sx, l.pz, l.py, l.px, l.KD, l.KH, l.KW);
        } else if (kch == 1) {
            dim3 grid(mtiles, ntiles / 4, B);
            k_conv<4, 1><<<grid, 32, 0, stream>>>(
                src, dst, wf,
                l.ID, l.IH, l.IW, l.OD, l.OH, l.OW, l.Coutpad, ntiles,
                l.sz, l.sy, l.sx, l.pz, l.py, l.px, l.KD, l.KH, l.KW);
        } else {
            dim3 grid(mtiles, ntiles / 4, B);
            k_conv<4, 2><<<grid, 32, 0, stream>>>(
                src, dst, wf,
                l.ID, l.IH, l.IW, l.OD, l.OH, l.OW, l.Coutpad, ntiles,
                l.sz, l.sy, l.sx, l.pz, l.py, l.px, l.KD, l.KH, l.KW);
        }

        _Float16* tmp = src; src = dst; dst = tmp;
    }

    // `src` now holds conv_out activations: [B, 400, 4096] channel-last.
    const long feats = (long)B * 400 * 4096;
    k_final<<<B * 400, 256, 0, stream>>>(
        src, (__hip_bfloat16*)d_out, counters, 4096, 400, feats, (long)out_size);
    k_offsets<<<1, 32, 0, stream>>>(
        counters, (__hip_bfloat16*)d_out, feats + (long)B * 400, B, (long)out_size);
}
